// AdaLeRobotEnhancedPerceptionModel_40724879901030
// MI455X (gfx1250) — compile-verified, matrix operand order checked by reading, not run on hardware
//
#include <hip/hip_runtime.h>
#include <math.h>

// ---------------------------------------------------------------------------
// Problem constants (from the reference)
// ---------------------------------------------------------------------------
#define LYR 8
#define BATCH 32
#define SEQ 64
#define DMODEL 1024
#define NHEAD 16
#define HDIM 64
#define DFF 4096
#define DFF2 2048
#define MROWS (BATCH * SEQ)      // 2048
#define NREL 63                  // 2*MAXREL-1

// vector types
typedef __attribute__((ext_vector_type(16))) _Float16 v16h;
typedef __attribute__((ext_vector_type(8)))  float    v8f;
typedef __attribute__((ext_vector_type(4)))  unsigned v4u;
typedef __attribute__((ext_vector_type(8)))  int      v8i;
typedef __attribute__((ext_vector_type(4)))  int      v4i;

__device__ __forceinline__ float gelu_exact(float x) {
    return 0.5f * x * (1.0f + erff(x * 0.70710678118654752f));
}
__device__ __forceinline__ float sigmoid_f(float x) {
    return 1.0f / (1.0f + expf(-x));
}

// ---------------------------------------------------------------------------
// TDM descriptor build + issue: load a 128(rows) x 32(k) f16 tile whose rows
// are contiguous-in-K (matrix is [R][K] f16 row-major) into LDS with a 1-dword
// pad after every 16 dwords  ->  LDS row stride of 34 halves.
//   group1 flags: data_size=2B, pad_enable, pad_interval=3 (16 dwords),
//                 pad_amount=0 (1 dword)
// ---------------------------------------------------------------------------
__device__ __forceinline__ void tdm_load_tile_f16(const _Float16* gptr, unsigned lds_byte,
                                                  int row0, int k0, int K, int R)
{
    unsigned long long ga = (unsigned long long)(size_t)gptr +
                            2ull * ((size_t)(unsigned)row0 * (unsigned)K + (unsigned)k0);
    v4u g0;
    g0[0] = 1u;                                   // count=1, is_restore=0
    g0[1] = lds_byte;                             // lds_addr
    g0[2] = (unsigned)ga;                         // global_addr[31:0]
    g0[3] = (unsigned)((ga >> 32) & 0x01FFFFFFu) | 0x80000000u; // ga[56:32] | type=2

    v8i g1;
    g1[0] = (int)0x00D10000u;                     // data_size=1(2B), pad_en, pad_int=3
    g1[1] = (int)(((unsigned)K & 0xFFFFu) << 16); // tensor_dim0[15:0] (atomic addr=0)
    g1[2] = (int)((((unsigned)K >> 16) & 0xFFFFu) | (((unsigned)R & 0xFFFFu) << 16));
    g1[3] = (int)((((unsigned)R >> 16) & 0xFFFFu) | (32u << 16)); // tile_dim0=32
    g1[4] = 128;                                  // tile_dim1=128, tile_dim2=0
    g1[5] = K;                                    // tensor_dim0_stride[31:0]
    g1[6] = 0;                                    // stride[47:32]=0, dim1_stride lo=0
    g1[7] = 0;

    v4i g2; g2[0] = 1; g2[1] = 0; g2[2] = 0; g2[3] = 0;        // tensor_dim2=1
    v4i g3; g3[0] = 0; g3[1] = (int)(1u << 16); g3[2] = 0; g3[3] = 0; // tensor_dim4=1
    v8i g4; g4[0] = 0; g4[1] = 0; g4[2] = 0; g4[3] = 0;
            g4[4] = 0; g4[5] = 0; g4[6] = 0; g4[7] = 0;        // unused extra group

    __builtin_amdgcn_tensor_load_to_lds(g0, g1, g2, g3, g4, 0);
}

// ---------------------------------------------------------------------------
// Tiled WMMA GEMM (f16 inputs, fp32 accumulate):
//   Cf[M,N] / Ch[M,N] = act( A16[M,K] @ W16T[N,K]^T + bias[N] (+ resid[M,N]) )
// BM=BN=128, BK=32, 256 threads (8 waves); wave -> 32x64 patch (8 WMMA tiles).
// Both tiles staged by the Tensor Data Mover into double-buffered LDS.
// ---------------------------------------------------------------------------
#define BM 128
#define BN 128
#define BK 32
#define LSTR 34   // halves per LDS row (32 data + 1 dword TDM pad)

__global__ __launch_bounds__(256)
void gemm_wmma_kernel(const _Float16* __restrict__ A16, const _Float16* __restrict__ W16T,
                      const float* __restrict__ bias, const float* __restrict__ resid,
                      float* __restrict__ Cf, _Float16* __restrict__ Ch,
                      int M, int N, int K, int act)
{
    __shared__ _Float16 Alds[2][BM * LSTR];
    __shared__ _Float16 Blds[2][BN * LSTR];

    const int tid  = threadIdx.x;
    const int wave = tid >> 5;
    const int lane = tid & 31;
    const int wm   = wave >> 1;          // 0..3 -> 32-row strip
    const int wn   = wave & 1;           // 0..1 -> 64-col strip
    const int l15  = lane & 15;
    const int kg   = lane >> 4;
    const int bm0  = blockIdx.y * BM;
    const int bn0  = blockIdx.x * BN;

    // fragment k-pair offsets within the 32-wide slab
    int kp[8];
#pragma unroll
    for (int p = 0; p < 8; ++p)
        kp[p] = (p < 4) ? (kg * 8 + 2 * p) : (16 + kg * 8 + 2 * (p - 4));

    v8f acc[2][4];
    const v8f vzero = {0.f, 0.f, 0.f, 0.f, 0.f, 0.f, 0.f, 0.f};
#pragma unroll
    for (int a = 0; a < 2; ++a)
#pragma unroll
        for (int b = 0; b < 4; ++b) acc[a][b] = vzero;

    typedef union { v16h v; unsigned u[8]; } frag_u;

    const unsigned aoff0 = (unsigned)(size_t)&Alds[0][0];
    const unsigned aoff1 = (unsigned)(size_t)&Alds[1][0];
    const unsigned boff0 = (unsigned)(size_t)&Blds[0][0];
    const unsigned boff1 = (unsigned)(size_t)&Blds[1][0];

    const int nk = K / BK;

    if (wave == 0) {  // wave-uniform: wave 0 drives the TDM pipeline
        tdm_load_tile_f16(A16, aoff0, bm0, 0, K, M);
        tdm_load_tile_f16(W16T, boff0, bn0, 0, K, N);
    }

    for (int it = 0; it < nk; ++it) {
        const int buf = it & 1;
        const _Float16* ab = &Alds[buf][0];
        const _Float16* bb = &Blds[buf][0];

        if (wave == 0) {
            if (it + 1 < nk) {
                const unsigned na = (buf ? aoff0 : aoff1);
                const unsigned nb = (buf ? boff0 : boff1);
                tdm_load_tile_f16(A16, na, bm0, (it + 1) * BK, K, M);
                tdm_load_tile_f16(W16T, nb, bn0, (it + 1) * BK, K, N);
                __builtin_amdgcn_s_wait_tensorcnt(2);   // current pair done
            } else {
                __builtin_amdgcn_s_wait_tensorcnt(0);
            }
        }
        __syncthreads();   // publish TDM-written LDS to all waves

        // ---- fragments from LDS ----
        v16h af[2];
#pragma unroll
        for (int a = 0; a < 2; ++a) {
            int m = wm * 32 + a * 16 + l15;
            frag_u t;
#pragma unroll
            for (int p = 0; p < 8; ++p)
                t.u[p] = *(const unsigned*)(&ab[m * LSTR + kp[p]]);
            af[a] = t.v;
        }
        v16h bf[4];
#pragma unroll
        for (int b = 0; b < 4; ++b) {
            int n = wn * 64 + b * 16 + l15;
            frag_u t;
#pragma unroll
            for (int p = 0; p < 8; ++p)
                t.u[p] = *(const unsigned*)(&bb[n * LSTR + kp[p]]);
            bf[b] = t.v;
        }

        // ---- 8 WMMAs, fp32 accumulate ----
#pragma unroll
        for (int a = 0; a < 2; ++a)
#pragma unroll
            for (int b = 0; b < 4; ++b)
                acc[a][b] = __builtin_amdgcn_wmma_f32_16x16x32_f16(
                    false, af[a], false, bf[b], (short)0, acc[a][b], false, false);

        __syncthreads();   // compute done before next TDM overwrites buf^1
    }

    // ---- epilogue: bias (+resid) (+gelu), fp32 and/or f16 outputs ----
#pragma unroll
    for (int a = 0; a < 2; ++a) {
#pragma unroll
        for (int b = 0; b < 4; ++b) {
            int ncol  = bn0 + wn * 64 + b * 16 + l15;
            int mbase = bm0 + wm * 32 + a * 16 + 8 * kg;
            float bv = bias[ncol];
#pragma unroll
            for (int r = 0; r < 8; ++r) {
                int mr = mbase + r;
                float val = acc[a][b][r] + bv;
                if (resid) val += resid[(size_t)mr * N + ncol];
                if (act == 1) val = gelu_exact(val);
                if (Cf) Cf[(size_t)mr * N + ncol] = val;
                if (Ch) Ch[(size_t)mr * N + ncol] = (_Float16)val;
            }
        }
    }
}

// ---------------------------------------------------------------------------
// Weight convert + transpose: W fp32 [K][N] -> W16T f16 [N][K]
// ---------------------------------------------------------------------------
__global__ __launch_bounds__(256)
void cvtT_kernel(const float* __restrict__ W, _Float16* __restrict__ WT, int K, int N)
{
    __shared__ float t[32][33];
    const int n0 = blockIdx.x * 32;
    const int k0 = blockIdx.y * 32;
    const int tx = threadIdx.x & 31;
    const int ty = threadIdx.x >> 5;      // 0..7
#pragma unroll
    for (int j = 0; j < 4; ++j)
        t[ty + j * 8][tx] = W[(size_t)(k0 + ty + j * 8) * N + n0 + tx];
    __syncthreads();
#pragma unroll
    for (int j = 0; j < 4; ++j)
        WT[(size_t)(n0 + ty + j * 8) * K + k0 + tx] = (_Float16)t[tx][ty + j * 8];
}

// ---------------------------------------------------------------------------
// h = x (fp32)  and  h16 = f16(x)
// ---------------------------------------------------------------------------
__global__ void init_h_kernel(const float* __restrict__ x, float* __restrict__ h,
                              _Float16* __restrict__ h16, int n)
{
    int i = blockIdx.x * 256 + threadIdx.x;
    if (i < n) { float v = x[i]; h[i] = v; h16[i] = (_Float16)v; }
}

// ---------------------------------------------------------------------------
// Per-layer relative-position bias: relmean[r] = mean_d rel[r][d]
// ---------------------------------------------------------------------------
__global__ void relmean_kernel(const float* __restrict__ rel, float* __restrict__ rm)
{
    int r = threadIdx.x;
    if (r < NREL) {
        float s = 0.f;
        for (int d = 0; d < HDIM; ++d) s += rel[r * HDIM + d];
        rm[r] = s * (1.0f / (float)HDIM);
    }
}

// ---------------------------------------------------------------------------
// Attention: one block per (batch, head); exact fp32; writes f16 context.
// ---------------------------------------------------------------------------
__global__ __launch_bounds__(256)
void attn_kernel(const float* __restrict__ q, const float* __restrict__ k,
                 const float* __restrict__ v, const float* __restrict__ rm,
                 _Float16* __restrict__ o16)
{
    __shared__ float qs[SEQ][HDIM + 1];   // reused as probs after scores
    __shared__ float ks[SEQ][HDIM + 1];
    __shared__ float vs[SEQ][HDIM + 1];

    const int b   = blockIdx.x >> 4;
    const int hh  = blockIdx.x & 15;
    const int tid = threadIdx.x;
    const size_t rowbase = (size_t)b * SEQ;
    const int colbase = hh * HDIM;

#pragma unroll
    for (int i = 0; i < 16; ++i) {
        int id = tid + i * 256;
        int r = id >> 6, c = id & 63;
        size_t g = (rowbase + r) * DMODEL + colbase + c;
        qs[r][c] = q[g];
        ks[r][c] = k[g];
        vs[r][c] = v[g];
    }
    __syncthreads();

    float sc[16];
#pragma unroll
    for (int t = 0; t < 16; ++t) {
        int id = tid + t * 256;
        int i = id >> 6, j = id & 63;
        float acc = 0.f;
        for (int d = 0; d < HDIM; ++d) acc += qs[i][d] * ks[j][d];
        int rp = i - j + 31;
        rp = rp < 0 ? 0 : (rp > 62 ? 62 : rp);
        sc[t] = acc * 0.125f + rm[rp];
    }
    __syncthreads();
#pragma unroll
    for (int t = 0; t < 16; ++t) {
        int id = tid + t * 256;
        qs[id >> 6][id & 63] = sc[t];
    }
    __syncthreads();

    if (tid < SEQ) {
        int i = tid;
        float mx = -3.0e38f;
        for (int j = 0; j < SEQ; ++j) mx = fmaxf(mx, qs[i][j]);
        float s = 0.f;
        for (int j = 0; j < SEQ; ++j) { float e = expf(qs[i][j] - mx); qs[i][j] = e; s += e; }
        float inv = 1.0f / s;
        for (int j = 0; j < SEQ; ++j) qs[i][j] *= inv;
    }
    __syncthreads();

#pragma unroll
    for (int t = 0; t < 16; ++t) {
        int id = tid + t * 256;
        int i = id >> 6, d = id & 63;
        float acc = 0.f;
        for (int j = 0; j < SEQ; ++j) acc += qs[i][j] * vs[j][d];
        o16[(rowbase + i) * DMODEL + colbase + d] = (_Float16)acc;
    }
}

// ---------------------------------------------------------------------------
// Gated FFN elementwise: u1_16 = f16( gelu( up * sigmoid(gate) ) )
// ---------------------------------------------------------------------------
__global__ void gated_kernel(const float* __restrict__ up, const float* __restrict__ gate,
                             _Float16* __restrict__ out16, int n)
{
    int i = blockIdx.x * 256 + threadIdx.x;
    if (i < n) {
        float x = up[i] * sigmoid_f(gate[i]);
        out16[i] = (_Float16)gelu_exact(x);
    }
}

// ---------------------------------------------------------------------------
// LayerNorm over rows of 1024; writes fp32 and f16 outputs
// ---------------------------------------------------------------------------
__global__ __launch_bounds__(256)
void layernorm_kernel(const float* __restrict__ in, const float* __restrict__ g,
                      const float* __restrict__ be, float* __restrict__ outf,
                      _Float16* __restrict__ outh)
{
    __shared__ float red[256];
    const int row = blockIdx.x;
    const int tid = threadIdx.x;
    const size_t base = (size_t)row * DMODEL;

    float v[4];
    float s = 0.f;
#pragma unroll
    for (int i = 0; i < 4; ++i) { v[i] = in[base + tid + i * 256]; s += v[i]; }
    red[tid] = s; __syncthreads();
    for (int st = 128; st > 0; st >>= 1) { if (tid < st) red[tid] += red[tid + st]; __syncthreads(); }
    float mean = red[0] * (1.0f / (float)DMODEL);
    __syncthreads();

    float s2 = 0.f;
#pragma unroll
    for (int i = 0; i < 4; ++i) { float d = v[i] - mean; s2 += d * d; }
    red[tid] = s2; __syncthreads();
    for (int st = 128; st > 0; st >>= 1) { if (tid < st) red[tid] += red[tid + st]; __syncthreads(); }
    float rs = rsqrtf(red[0] * (1.0f / (float)DMODEL) + 1e-5f);

#pragma unroll
    for (int i = 0; i < 4; ++i) {
        int c = tid + i * 256;
        float val = (v[i] - mean) * rs * g[c] + be[c];
        outf[base + c] = val;
        outh[base + c] = (_Float16)val;
    }
}

// ---------------------------------------------------------------------------
// Launch orchestration
// ---------------------------------------------------------------------------
static inline void launch_gemm(const _Float16* A16, const _Float16* W16T,
                               const float* bias, const float* resid,
                               float* Cf, _Float16* Ch,
                               int M, int N, int K, int act, hipStream_t s)
{
    dim3 grid(N / BN, M / BM);
    gemm_wmma_kernel<<<grid, 256, 0, s>>>(A16, W16T, bias, resid, Cf, Ch, M, N, K, act);
}

static inline void launch_cvtT(const float* W, _Float16* WT, int K, int N, hipStream_t s)
{
    dim3 grid(N / 32, K / 32);
    cvtT_kernel<<<grid, 256, 0, s>>>(W, WT, K, N);
}

extern "C" void kernel_launch(void* const* d_in, const int* in_sizes, int n_in,
                              void* d_out, int out_size, void* d_ws, size_t ws_size,
                              hipStream_t stream)
{
    const float* x   = (const float*)d_in[0];
    const float* Wq  = (const float*)d_in[1];
    const float* bq  = (const float*)d_in[2];
    const float* Wk  = (const float*)d_in[3];
    const float* bk  = (const float*)d_in[4];
    const float* Wv  = (const float*)d_in[5];
    const float* bv  = (const float*)d_in[6];
    const float* rel = (const float*)d_in[7];
    const float* Wo  = (const float*)d_in[8];
    const float* bo  = (const float*)d_in[9];
    const float* Wg  = (const float*)d_in[10];
    const float* bg  = (const float*)d_in[11];
    const float* W1  = (const float*)d_in[12];
    const float* b1  = (const float*)d_in[13];
    const float* W2  = (const float*)d_in[14];
    const float* b2  = (const float*)d_in[15];
    const float* W3  = (const float*)d_in[16];
    const float* b3  = (const float*)d_in[17];
    const float* lng = (const float*)d_in[18];
    const float* lnb = (const float*)d_in[19];

    // ---- workspace layout ----
    char* wsb = (char*)d_ws;
    size_t off = 0;
    auto allocF = [&](size_t n) { float* p = (float*)(wsb + off); off += n * 4; return p; };
    auto allocH = [&](size_t n) { _Float16* p = (_Float16*)(wsb + off); off += n * 2; return p; };

    float* h    = allocF((size_t)MROWS * DMODEL);
    float* hsum = allocF((size_t)MROWS * DMODEL);
    float* qb   = allocF((size_t)MROWS * DMODEL);
    float* kb   = allocF((size_t)MROWS * DMODEL);
    float* vb   = allocF((size_t)MROWS * DMODEL);
    float* tg   = allocF((size_t)MROWS * DFF);
    float* tu   = allocF((size_t)MROWS * DFF);
    float* rm   = allocF(64);

    _Float16* h16  = allocH((size_t)MROWS * DMODEL);
    _Float16* o16  = allocH((size_t)MROWS * DMODEL);
    _Float16* u116 = allocH((size_t)MROWS * DFF);
    _Float16* t216 = allocH((size_t)MROWS * DFF2);

    _Float16* wq16 = allocH((size_t)DMODEL * DMODEL);
    _Float16* wk16 = allocH((size_t)DMODEL * DMODEL);
    _Float16* wv16 = allocH((size_t)DMODEL * DMODEL);
    _Float16* wo16 = allocH((size_t)DMODEL * DMODEL);
    _Float16* wg16 = allocH((size_t)DMODEL * DFF);
    _Float16* w116 = allocH((size_t)DMODEL * DFF);
    _Float16* w216 = allocH((size_t)DFF * DFF2);
    _Float16* w316 = allocH((size_t)DFF2 * DMODEL);

    const int nD  = MROWS * DMODEL;
    const int nFF = MROWS * DFF;

    init_h_kernel<<<(nD + 255) / 256, 256, 0, stream>>>(x, h, h16, nD);

    for (int l = 0; l < LYR; ++l) {
        const size_t oDD  = (size_t)l * DMODEL * DMODEL;
        const size_t oD   = (size_t)l * DMODEL;
        const size_t oDF  = (size_t)l * DMODEL * DFF;
        const size_t oF   = (size_t)l * DFF;
        const size_t oFF2 = (size_t)l * DFF * DFF2;
        const size_t oF2  = (size_t)l * DFF2;
        const size_t oF2D = (size_t)l * DFF2 * DMODEL;
        const size_t oR   = (size_t)l * NREL * HDIM;

        // f16 transposed weights for this layer
        launch_cvtT(Wq + oDD,  wq16, DMODEL, DMODEL, stream);
        launch_cvtT(Wk + oDD,  wk16, DMODEL, DMODEL, stream);
        launch_cvtT(Wv + oDD,  wv16, DMODEL, DMODEL, stream);
        launch_cvtT(Wo + oDD,  wo16, DMODEL, DMODEL, stream);
        launch_cvtT(Wg + oDF,  wg16, DMODEL, DFF,    stream);
        launch_cvtT(W1 + oDF,  w116, DMODEL, DFF,    stream);
        launch_cvtT(W2 + oFF2, w216, DFF,    DFF2,   stream);
        launch_cvtT(W3 + oF2D, w316, DFF2,   DMODEL, stream);

        // Q,K,V projections (WMMA + TDM)
        launch_gemm(h16, wq16, bq + oD, nullptr, qb, nullptr, MROWS, DMODEL, DMODEL, 0, stream);
        launch_gemm(h16, wk16, bk + oD, nullptr, kb, nullptr, MROWS, DMODEL, DMODEL, 0, stream);
        launch_gemm(h16, wv16, bv + oD, nullptr, vb, nullptr, MROWS, DMODEL, DMODEL, 0, stream);

        relmean_kernel<<<1, 64, 0, stream>>>(rel + oR, rm);
        attn_kernel<<<BATCH * NHEAD, 256, 0, stream>>>(qb, kb, vb, rm, o16);

        // h = h + o @ Wo + bo  (fp32 + f16 shadow)
        launch_gemm(o16, wo16, bo + oD, h, h, h16, MROWS, DMODEL, DMODEL, 0, stream);

        // gated FFN
        launch_gemm(h16, wg16, bg + oF, nullptr, tg, nullptr, MROWS, DFF, DMODEL, 0, stream);
        launch_gemm(h16, w116, b1 + oF, nullptr, tu, nullptr, MROWS, DFF, DMODEL, 0, stream);
        gated_kernel<<<(nFF + 255) / 256, 256, 0, stream>>>(tu, tg, u116, nFF);
        launch_gemm(u116, w216, b2 + oF2, nullptr, nullptr, t216, MROWS, DFF2, DFF, 1, stream);
        launch_gemm(t216, w316, b3 + oD, h, hsum, nullptr, MROWS, DMODEL, DFF2, 0, stream);

        // h = LN(hsum); final layer writes fp32 straight to d_out
        float* outp = (l == LYR - 1) ? (float*)d_out : h;
        layernorm_kernel<<<MROWS, 256, 0, stream>>>(hsum, lng + oD, lnb + oD, outp, h16);
    }
    (void)in_sizes; (void)n_in; (void)out_size; (void)ws_size;
}
